// MultiHeadAttention_24902220382931
// MI455X (gfx1250) — compile-verified, matrix operand order checked by reading, not run on hardware
//
#include <hip/hip_runtime.h>
#include <hip/hip_fp16.h>

#define T_TOK 12160
#define DMODEL 1024
#define NHEAD 8
#define DKK 128

typedef __attribute__((ext_vector_type(16))) _Float16 v16h;
typedef __attribute__((ext_vector_type(8)))  float    v8f;

__device__ __forceinline__ v8f vzero() {
  v8f z = {0.f,0.f,0.f,0.f,0.f,0.f,0.f,0.f};
  return z;
}

__device__ __forceinline__ v8f wmma16(v16h a, v16h b, v8f c) {
  return __builtin_amdgcn_wmma_f32_16x16x32_f16(false, a, false, b, (short)0, c, false, false);
}

// 16-bit A/B fragment: lane half=l>>4 selects K chunks; base points at K=0 of this
// lane's row. half=0 -> elements [0,8) and [16,24); half=1 -> [8,16) and [24,32).
__device__ __forceinline__ v16h ld_frag(const _Float16* base, int half) {
  union { v16h v; uint4 q[2]; } f;
  const _Float16* p = base + (half ? 8 : 0);
  f.q[0] = *(const uint4*)(p);
  f.q[1] = *(const uint4*)(p + 16);
  return f.v;
}

// CDNA5 async copy: one 16B chunk per lane, global -> LDS, tracked by ASYNCcnt.
__device__ __forceinline__ void async_ld16(unsigned lds_off, const _Float16* g) {
  asm volatile("global_load_async_to_lds_b128 %0, %1, off"
               :: "v"(lds_off), "v"((unsigned long long)(size_t)g)
               : "memory");
}
__device__ __forceinline__ void wait_async0() {
  asm volatile("s_wait_asynccnt 0x0" ::: "memory");
}

// ---------------- conversion kernels ----------------
__global__ void cvt_kernel(const float* __restrict__ src, _Float16* __restrict__ dst, int n) {
  int i = blockIdx.x * 256 + threadIdx.x;
  if (i < n) dst[i] = (_Float16)src[i];
}

// (8,512,64) f32 -> (8,64,512) f16 (transpose d<->n so B-frags are contiguous)
__global__ void wtr_kernel(const float* __restrict__ src, _Float16* __restrict__ dst) {
  int i = blockIdx.x * 256 + threadIdx.x;     // 8*512*64 = 262144 total
  int h = i >> 15, rem = i & 32767, d = rem >> 6, n = rem & 63;
  dst[(h << 15) + (n << 9) + d] = (_Float16)src[i];
}

// ---------------- QKV projection (register double-buffered) ----------------
// grid(95, 8): 128-token tile x head. 8 waves = 4(M of 32 rows) x 2(N half of 64 cols).
__global__ __launch_bounds__(256) void qkv_kernel(
    const _Float16* __restrict__ inp_h,
    const _Float16* __restrict__ w1T,   // [8][64][512] f16
    const _Float16* __restrict__ w2T,
    _Float16* __restrict__ dst, int transposed)
{
  int t0 = blockIdx.x * 128;
  int h  = blockIdx.y;
  int tid = threadIdx.x, wave = tid >> 5, lane = tid & 31, l16 = lane & 15, half = lane >> 4;
  int wm = wave & 3, wn = wave >> 2;
  const _Float16* wT = (wn ? w2T : w1T) + (size_t)h * (64 * 512);
  int d0 = wn * 512;
  const _Float16* abase = inp_h + (size_t)(t0 + wm * 32 + l16) * DMODEL + d0;

  v8f acc[2][4];
#pragma unroll
  for (int ms = 0; ms < 2; ++ms)
#pragma unroll
    for (int ns = 0; ns < 4; ++ns) acc[ms][ns] = vzero();

  v16h a0[2], b0[4];
#pragma unroll
  for (int ms = 0; ms < 2; ++ms) a0[ms] = ld_frag(abase + ms * 16 * DMODEL, half);
#pragma unroll
  for (int ns = 0; ns < 4; ++ns) b0[ns] = ld_frag(wT + (size_t)(ns * 16 + l16) * 512, half);

#pragma unroll
  for (int ks = 0; ks < 16; ++ks) {
    int kn = (ks < 15) ? ks + 1 : 15;
    v16h a1[2], b1[4];
#pragma unroll
    for (int ms = 0; ms < 2; ++ms) {
      const _Float16* ap = abase + ms * 16 * DMODEL + kn * 32;
      a1[ms] = ld_frag(ap, half);
      __builtin_prefetch(ap + 32, 0, 1);
    }
#pragma unroll
    for (int ns = 0; ns < 4; ++ns)
      b1[ns] = ld_frag(wT + (size_t)(ns * 16 + l16) * 512 + kn * 32, half);
#pragma unroll
    for (int ns = 0; ns < 4; ++ns) {
      acc[0][ns] = wmma16(a0[0], b0[ns], acc[0][ns]);
      acc[1][ns] = wmma16(a0[1], b0[ns], acc[1][ns]);
    }
#pragma unroll
    for (int ms = 0; ms < 2; ++ms) a0[ms] = a1[ms];
#pragma unroll
    for (int ns = 0; ns < 4; ++ns) b0[ns] = b1[ns];
  }

  if (!transposed) {                                      // q,k: [h][t][128]
    _Float16* d = dst + (size_t)h * T_TOK * DKK;
#pragma unroll
    for (int ms = 0; ms < 2; ++ms)
#pragma unroll
      for (int ns = 0; ns < 4; ++ns)
#pragma unroll
        for (int r = 0; r < 8; ++r) {
          int t = t0 + wm * 32 + ms * 16 + r + 8 * half;
          int col = wn * 64 + ns * 16 + l16;
          d[(size_t)t * DKK + col] = (_Float16)acc[ms][ns][r];
        }
  } else {                                                // v: transposed [h][dv][T]
    _Float16* d = dst + (size_t)h * DKK * T_TOK;
#pragma unroll
    for (int ms = 0; ms < 2; ++ms)
#pragma unroll
      for (int ns = 0; ns < 4; ++ns)
#pragma unroll
        for (int r = 0; r < 8; ++r) {
          int t = t0 + wm * 32 + ms * 16 + r + 8 * half;
          int col = wn * 64 + ns * 16 + l16;
          d[(size_t)col * T_TOK + t] = (_Float16)acc[ms][ns][r];
        }
  }
}

// ---------------- flash attention with async-to-LDS K/V staging ----------------
// grid(8 heads, 32 batches, 4 q-blocks of 128). Each wave owns 16 q rows; online softmax.
// K tile (32x128) and V^T tile (128x32) staged to LDS once per k-block via
// global_load_async_to_lds_b128 (shared by all 8 waves instead of 8x redundant loads).
__global__ __launch_bounds__(256) void attn_kernel(
    const _Float16* __restrict__ qh, const _Float16* __restrict__ kh,
    const _Float16* __restrict__ vT, float* __restrict__ ot)
{
  __shared__ _Float16 kbuf[32][128];                      // [kpos_local][d]
  __shared__ _Float16 vbuf[128][32];                      // [dv][kpos_local]
  __shared__ _Float16 pbuf[8][16][32];                    // per-wave P re-layout bounce
  int h = blockIdx.x, b = blockIdx.y, qb = blockIdx.z;
  int L = 256 + 8 * b;
  int off = 256 * b + 4 * b * (b - 1);
  int q0 = qb * 128;
  if (q0 >= L) return;
  int tid = threadIdx.x, wave = tid >> 5, lane = tid & 31, l16 = lane & 15, half = lane >> 4;

  int qrow = q0 + wave * 16 + l16;
  int qcl = qrow < L ? qrow : L - 1;
  const _Float16* qptr = qh + ((size_t)h * T_TOK + off + qcl) * DKK;
  v16h qa[4];
#pragma unroll
  for (int kk = 0; kk < 4; ++kk) qa[kk] = ld_frag(qptr + kk * 32, half);

  const _Float16* kbase = kh + (size_t)h * T_TOK * DKK;
  const _Float16* vbase = vT + (size_t)h * DKK * T_TOK;

  v8f acc[8];
#pragma unroll
  for (int dt = 0; dt < 8; ++dt) acc[dt] = vzero();
  float mrun[8], lrun[8];
#pragma unroll
  for (int r = 0; r < 8; ++r) { mrun[r] = -1e30f; lrun[r] = 0.f; }

  int nkb = (L + 31) >> 5;
  for (int kb = 0; kb < nkb; ++kb) {
    int t0k = off + kb * 32;
    // ---- stage K tile: 512 chunks of 16B, 2 per thread ----
#pragma unroll
    for (int u = 0; u < 2; ++u) {
      int c = tid * 2 + u;
      int row = c >> 4, col = (c & 15) << 3;
      int kp = t0k + row; if (kp > T_TOK - 1) kp = T_TOK - 1;
      async_ld16((unsigned)(size_t)&kbuf[row][col], kbase + (size_t)kp * DKK + col);
    }
    // ---- stage V^T tile: 512 chunks of 16B, 2 per thread ----
#pragma unroll
    for (int u = 0; u < 2; ++u) {
      int c = tid * 2 + u;
      int dv = c >> 2, tc = (c & 3) << 3;
      int ts = t0k + tc; if (ts > T_TOK - 8) ts = T_TOK - 8;
      async_ld16((unsigned)(size_t)&vbuf[dv][tc], vbase + (size_t)dv * T_TOK + ts);
    }
    wait_async0();
    __syncthreads();

    // ---- S = Q . K^T for 32 key positions (2 N-tiles) from LDS ----
    v8f s0 = vzero(), s1 = vzero();
#pragma unroll
    for (int kk = 0; kk < 4; ++kk) {
      s0 = wmma16(qa[kk], ld_frag(&kbuf[l16][kk * 32], half), s0);
      s1 = wmma16(qa[kk], ld_frag(&kbuf[16 + l16][kk * 32], half), s1);
    }
    const float sc = 0.03125f;                            // 1/sqrt(1024)
    int c0 = kb * 32 + l16, c1 = kb * 32 + 16 + l16;
    float p0[8], p1v[8], bm[8];
#pragma unroll
    for (int r = 0; r < 8; ++r) {
      float a0 = (c0 < L) ? s0[r] * sc : -1e30f;
      float a1 = (c1 < L) ? s1[r] * sc : -1e30f;
      p0[r] = a0; p1v[r] = a1; bm[r] = fmaxf(a0, a1);
    }
#pragma unroll
    for (int m = 1; m < 16; m <<= 1)
#pragma unroll
      for (int r = 0; r < 8; ++r) bm[r] = fmaxf(bm[r], __shfl_xor(bm[r], m, 32));
    float alpha[8], rs[8];
#pragma unroll
    for (int r = 0; r < 8; ++r) {
      float mn = fmaxf(mrun[r], bm[r]);
      alpha[r] = __expf(mrun[r] - mn);
      mrun[r] = mn;
      float e0 = __expf(p0[r] - mn), e1 = __expf(p1v[r] - mn);
      p0[r] = e0; p1v[r] = e1; rs[r] = e0 + e1;
    }
#pragma unroll
    for (int m = 1; m < 16; m <<= 1)
#pragma unroll
      for (int r = 0; r < 8; ++r) rs[r] += __shfl_xor(rs[r], m, 32);
#pragma unroll
    for (int r = 0; r < 8; ++r) lrun[r] = lrun[r] * alpha[r] + rs[r];
#pragma unroll
    for (int dt = 0; dt < 8; ++dt)
#pragma unroll
      for (int r = 0; r < 8; ++r) acc[dt][r] *= alpha[r];

    // P (16x32 f16) -> LDS -> A fragment
    _Float16* pw = &pbuf[wave][0][0];
#pragma unroll
    for (int r = 0; r < 8; ++r) {
      pw[(r + 8 * half) * 32 + l16]      = (_Float16)p0[r];
      pw[(r + 8 * half) * 32 + 16 + l16] = (_Float16)p1v[r];
    }
    v16h pa = ld_frag(pw + l16 * 32, half);

    // O += P . V from LDS V^T tile
#pragma unroll
    for (int dt = 0; dt < 8; ++dt)
      acc[dt] = wmma16(pa, ld_frag(&vbuf[dt * 16 + l16][0], half), acc[dt]);

    __syncthreads();   // all waves done reading kbuf/vbuf before next overwrite
  }

  float inv[8];
#pragma unroll
  for (int r = 0; r < 8; ++r) inv[r] = 1.f / lrun[r];
#pragma unroll
  for (int dt = 0; dt < 8; ++dt)
#pragma unroll
    for (int r = 0; r < 8; ++r) {
      int qr = q0 + wave * 16 + r + 8 * half;
      if (qr < L)
        ot[((size_t)(off + qr)) * DMODEL + h * DKK + dt * 16 + l16] = acc[dt][r] * inv[r];
    }
}

// ---------------- output projections + residual (pipelined) ----------------
// grid(95, 4): 128 rows x 256 cols of z. g<2 -> o1@proj1^T; g>=2 -> o2@proj2^T.
__global__ __launch_bounds__(256) void proj_kernel(
    const float* __restrict__ ot, const _Float16* __restrict__ p1,
    const _Float16* __restrict__ p2, const float* __restrict__ inp,
    float* __restrict__ out)
{
  int t0 = blockIdx.x * 128;
  int g  = blockIdx.y;
  int tid = threadIdx.x, wave = tid >> 5, lane = tid & 31, l16 = lane & 15, half = lane >> 4;
  const _Float16* proj = (g < 2) ? p1 : p2;
  int colloc = (g & 1) * 256;
  int osel = (g < 2) ? 0 : 64;
  int trow = t0 + wave * 16 + l16;
  const float* orow = ot + (size_t)trow * DMODEL;

  v8f acc[16];
#pragma unroll
  for (int nt = 0; nt < 16; ++nt) acc[nt] = vzero();

  auto build_af = [&](int ks) -> v16h {
    union { v16h v; unsigned u[8]; } af;
#pragma unroll
    for (int jj = 0; jj < 8; ++jj) {
      int j = ks * 32 + ((jj >> 2) << 4) + half * 8 + ((jj & 3) << 1);
      const float* sp = orow + ((j >> 6) << 7) + osel + (j & 63);
      float2 f2 = *(const float2*)sp;
      union { unsigned u; _Float16 hh[2]; } pk;
      pk.hh[0] = (_Float16)f2.x; pk.hh[1] = (_Float16)f2.y;
      af.u[jj] = pk.u;
    }
    return af.v;
  };
  auto loadB = [&](v16h* bf, int ks, int grp) {
#pragma unroll
    for (int i = 0; i < 4; ++i)
      bf[i] = ld_frag(proj + (size_t)(colloc + (grp * 4 + i) * 16 + l16) * 512 + ks * 32, half);
  };

  v16h af = build_af(0);
  v16h afn = af;
  v16h bc[4];
  loadB(bc, 0, 0);

#pragma unroll
  for (int step = 0; step < 64; ++step) {
    int ks = step >> 2, grp = step & 3;
    v16h bn[4];
    if (step < 63) loadB(bn, (step + 1) >> 2, (step + 1) & 3);
    if (grp == 0 && ks < 15) afn = build_af(ks + 1);
#pragma unroll
    for (int i = 0; i < 4; ++i)
      acc[grp * 4 + i] = wmma16(af, bc[i], acc[grp * 4 + i]);
    if (grp == 3) af = afn;
    if (step < 63) {
#pragma unroll
      for (int i = 0; i < 4; ++i) bc[i] = bn[i];
    }
  }

#pragma unroll
  for (int nt = 0; nt < 16; ++nt)
#pragma unroll
    for (int r = 0; r < 8; ++r) {
      int t = t0 + wave * 16 + r + 8 * half;
      int col = g * 256 + nt * 16 + l16;
      out[(size_t)t * DMODEL + col] = acc[nt][r] + inp[(size_t)t * DMODEL + col];
    }
}

// ---------------- layernorm (in place on d_out) ----------------
__global__ __launch_bounds__(256) void ln_kernel(float* __restrict__ z,
    const float* __restrict__ a2, const float* __restrict__ b2)
{
  __shared__ float sh[8];
  int row = blockIdx.x, tid = threadIdx.x;
  float* zr = z + (size_t)row * DMODEL;
  float4 v = ((const float4*)zr)[tid];
  float s = v.x + v.y + v.z + v.w;
#pragma unroll
  for (int m = 16; m >= 1; m >>= 1) s += __shfl_xor(s, m, 32);
  if ((tid & 31) == 0) sh[tid >> 5] = s;
  __syncthreads();
  float mu = (sh[0]+sh[1]+sh[2]+sh[3]+sh[4]+sh[5]+sh[6]+sh[7]) * (1.f / 1024.f);
  __syncthreads();
  float dx = v.x - mu, dy = v.y - mu, dz = v.z - mu, dw = v.w - mu;
  float ss = dx*dx + dy*dy + dz*dz + dw*dw;
#pragma unroll
  for (int m = 16; m >= 1; m >>= 1) ss += __shfl_xor(ss, m, 32);
  if ((tid & 31) == 0) sh[tid >> 5] = ss;
  __syncthreads();
  float var = (sh[0]+sh[1]+sh[2]+sh[3]+sh[4]+sh[5]+sh[6]+sh[7]) * (1.f / 1023.f);
  float rinv = 1.f / (sqrtf(var) + 1e-3f);
  float4 av = ((const float4*)a2)[tid];
  float4 bv = ((const float4*)b2)[tid];
  float4 o;
  o.x = dx * rinv * av.x + bv.x;
  o.y = dy * rinv * av.y + bv.y;
  o.z = dz * rinv * av.z + bv.z;
  o.w = dw * rinv * av.w + bv.w;
  ((float4*)zr)[tid] = o;
}

extern "C" void kernel_launch(void* const* d_in, const int* in_sizes, int n_in,
                              void* d_out, int out_size, void* d_ws, size_t ws_size,
                              hipStream_t stream) {
  const float* inp   = (const float*)d_in[0];
  const float* w_qs1 = (const float*)d_in[1];
  const float* w_ks1 = (const float*)d_in[2];
  const float* w_vs1 = (const float*)d_in[3];
  const float* w_qs2 = (const float*)d_in[4];
  const float* w_ks2 = (const float*)d_in[5];
  const float* w_vs2 = (const float*)d_in[6];
  const float* proj1 = (const float*)d_in[7];
  const float* proj2 = (const float*)d_in[8];
  const float* a2    = (const float*)d_in[9];
  const float* b2    = (const float*)d_in[10];
  float* out = (float*)d_out;

  char* ws = (char*)d_ws;
  size_t o = 0;
  auto carve = [&](size_t bytes) -> char* {
    char* p = ws + o;
    o += (bytes + 255) & ~(size_t)255;
    return p;
  };
  _Float16* inp_h = (_Float16*)carve((size_t)T_TOK * DMODEL * 2);
  _Float16* wq1T  = (_Float16*)carve((size_t)8 * 64 * 512 * 2);
  _Float16* wk1T  = (_Float16*)carve((size_t)8 * 64 * 512 * 2);
  _Float16* wv1T  = (_Float16*)carve((size_t)8 * 64 * 512 * 2);
  _Float16* wq2T  = (_Float16*)carve((size_t)8 * 64 * 512 * 2);
  _Float16* wk2T  = (_Float16*)carve((size_t)8 * 64 * 512 * 2);
  _Float16* wv2T  = (_Float16*)carve((size_t)8 * 64 * 512 * 2);
  _Float16* p1h   = (_Float16*)carve((size_t)512 * 512 * 2);
  _Float16* p2h   = (_Float16*)carve((size_t)512 * 512 * 2);
  _Float16* qh    = (_Float16*)carve((size_t)8 * T_TOK * DKK * 2);
  _Float16* kh    = (_Float16*)carve((size_t)8 * T_TOK * DKK * 2);
  _Float16* vth   = (_Float16*)carve((size_t)8 * T_TOK * DKK * 2);
  float*    ot    = (float*)   carve((size_t)T_TOK * DMODEL * 4);

  int nInp = T_TOK * DMODEL;
  cvt_kernel<<<(nInp + 255) / 256, 256, 0, stream>>>(inp, inp_h, nInp);
  wtr_kernel<<<1024, 256, 0, stream>>>(w_qs1, wq1T);
  wtr_kernel<<<1024, 256, 0, stream>>>(w_ks1, wk1T);
  wtr_kernel<<<1024, 256, 0, stream>>>(w_vs1, wv1T);
  wtr_kernel<<<1024, 256, 0, stream>>>(w_qs2, wq2T);
  wtr_kernel<<<1024, 256, 0, stream>>>(w_ks2, wk2T);
  wtr_kernel<<<1024, 256, 0, stream>>>(w_vs2, wv2T);
  cvt_kernel<<<1024, 256, 0, stream>>>(proj1, p1h, 512 * 512);
  cvt_kernel<<<1024, 256, 0, stream>>>(proj2, p2h, 512 * 512);

  qkv_kernel<<<dim3(95, 8), 256, 0, stream>>>(inp_h, wq1T, wq2T, qh, 0);
  qkv_kernel<<<dim3(95, 8), 256, 0, stream>>>(inp_h, wk1T, wk2T, kh, 0);
  qkv_kernel<<<dim3(95, 8), 256, 0, stream>>>(inp_h, wv1T, wv2T, vth, 1);

  attn_kernel<<<dim3(8, 32, 4), 256, 0, stream>>>(qh, kh, vth, ot);

  proj_kernel<<<dim3(95, 4), 256, 0, stream>>>(ot, p1h, p2h, inp, out);

  ln_kernel<<<T_TOK, 256, 0, stream>>>(out, a2, b2);
}
